// Seq2SeqAttn_80427557585373
// MI455X (gfx1250) — compile-verified, matrix operand order checked by reading, not run on hardware
//
#include <hip/hip_runtime.h>

// ---------------------------------------------------------------------------
// Seq2Seq + Luong attention for MI455X (gfx1250, wave32, WMMA bf16).
// B=128, S=336, IN=64, H=1024, T=96, OUT=1, gates G=4H=4096.
// All GEMMs use v_wmma_f32_16x16x32_bf16; weights + encoder state kept bf16
// so the whole recurrent working set (~120MB) stays L2-resident (192MB L2).
// GEMM inner loop: double-buffered LDS weight staging, one barrier per
// K-step, global weight prefetch overlapped with the WMMA chain.
// ---------------------------------------------------------------------------

#define CB   128
#define CS   336
#define CIN  64
#define CH   1024
#define CT   96
#define CG   4096          // 4*H
#define LDSB_STRIDE 40     // halves per LDS row (80B): 16B aligned, conflict-free
#define LDSB_HALVES (64 * LDSB_STRIDE)

typedef __bf16 bf16;
typedef __attribute__((ext_vector_type(16))) __bf16 v16bf;
typedef __attribute__((ext_vector_type(8)))  float  v8f;
typedef __attribute__((ext_vector_type(4)))  unsigned int u32x4;

union BfBits { unsigned short s; bf16 b; };

__device__ __forceinline__ bf16 f2bf(float f) {
    union { float f; unsigned u; } v; v.f = f;
    unsigned r = v.u + 0x7FFFu + ((v.u >> 16) & 1u);   // round-to-nearest-even
    BfBits o; o.s = (unsigned short)(r >> 16);
    return o.b;
}
__device__ __forceinline__ float bf2f(bf16 x) {
    BfBits i; i.b = x;
    union { unsigned u; float f; } v; v.u = ((unsigned)i.s) << 16;
    return v.f;
}

// ---------------------------------------------------------------------------
// Setup / conversion kernels
// ---------------------------------------------------------------------------
__global__ void k_f2bf(bf16* __restrict__ dst, const float* __restrict__ src, int n) {
    int i = blockIdx.x * 256 + threadIdx.x;
    if (i < n) dst[i] = f2bf(src[i]);
}
__global__ void k_zero_f(float* __restrict__ p, int n) {
    int i = blockIdx.x * 256 + threadIdx.x;
    if (i < n) p[i] = 0.0f;
}
__global__ void k_zero_bf(bf16* __restrict__ p, int n) {
    int i = blockIdx.x * 256 + threadIdx.x;
    if (i < n) p[i] = f2bf(0.0f);
}
__global__ void k_bias_sum(float* __restrict__ d, const float* __restrict__ a,
                           const float* __restrict__ b, int n) {
    int i = blockIdx.x * 256 + threadIdx.x;
    if (i < n) d[i] = a[i] + b[i];
}
// dec_W_ih is [4096,1025]: column 0 -> fp32 rank-1 vector, cols 1..1024 -> bf16 GEMM weight
__global__ void k_pack_decwih(bf16* __restrict__ wctx, float* __restrict__ wcol0,
                              const float* __restrict__ W) {
    int i = blockIdx.x * 256 + threadIdx.x;
    if (i < CG * CH) {
        int j = i >> 10, k = i & (CH - 1);
        wctx[i] = f2bf(W[(size_t)j * (CH + 1) + 1 + k]);
    }
    if (i < CG) wcol0[i] = W[(size_t)i * (CH + 1)];
}

// ---------------------------------------------------------------------------
// WMMA GEMM:  C[128, N] = bias[n] (+ py[m]*wcol0[n]) + A1*W1^T + A2*W2^T
//   A*: [128, K] bf16 row-major (stride lda), W*: [N, K] bf16 row-major.
//   Block = 256 thr (8 waves) -> 128x64 output tile; wave w owns rows 16w..16w+15
//   and 4 accumulator fragments. 32x64 weight strip double-buffered in LDS;
//   next strip prefetched to registers while the WMMA chain runs.
// ---------------------------------------------------------------------------
__global__ __launch_bounds__(256) void gates_gemm(
    const bf16* __restrict__ A1, int lda1, int K1, const bf16* __restrict__ W1, int ldw1,
    const bf16* __restrict__ A2, int lda2, int K2, const bf16* __restrict__ W2, int ldw2,
    const float* __restrict__ bsum, const float* __restrict__ py,
    const float* __restrict__ wcol0, float* __restrict__ Cout, int ldc)
{
    __shared__ __align__(16) bf16 ldsB[2 * LDSB_HALVES];

    const int tid  = threadIdx.x;
    const int lane = tid & 31;
    const int wave = tid >> 5;
    const int half = lane >> 4;     // ISA 16-bit frag half select
    const int row  = lane & 15;     // A: M-row in tile / B: N-col in tile
    const int n0   = blockIdx.x * 64;
    const int m0   = wave * 16;

    // cooperative stage coordinates: thread -> (weight row, 16B K-chunk)
    const int jr = tid >> 2;        // 0..63
    const int ks = (tid & 3) * 8;   // 0,8,16,24 halves

    v8f acc[4];
    #pragma unroll
    for (int nt = 0; nt < 4; ++nt) {
        const int colg = n0 + nt * 16 + row;
        #pragma unroll
        for (int r = 0; r < 8; ++r) {
            const int m = m0 + r + 8 * half;     // C/D layout: VGPR r -> M=r / r+8
            float v = bsum ? bsum[colg] : 0.0f;
            if (py) v += py[m] * wcol0[colg];
            acc[nt][r] = v;
        }
    }

    #pragma unroll 1
    for (int phase = 0; phase < 2; ++phase) {
        const bf16* A   = phase ? A2 : A1;
        const bf16* W   = phase ? W2 : W1;
        const int   K   = phase ? K2 : K1;
        const int   lda = phase ? lda2 : lda1;
        const int   ldw = phase ? ldw2 : ldw1;
        if (K == 0 || A == nullptr) continue;
        const int niter = K >> 5;

        const bf16* wp = W + (size_t)(n0 + jr) * ldw + ks;          // += 32 / iter
        const bf16* ap = A + (size_t)(m0 + row) * lda + half * 8;   // += 32 / iter

        // prime buffer 0 (leading barrier protects buffer reuse across phases)
        __syncthreads();
        *(u32x4*)&ldsB[jr * LDSB_STRIDE + ks] = *(const u32x4*)wp;
        __syncthreads();

        for (int it = 0; it < niter; ++it) {
            // A fragment for K-slice it*32 (ISA 16-bit A layout)
            union { u32x4 u[2]; v16bf v; } ua;
            ua.u[0] = *(const u32x4*)(ap);          // K = half*8 .. +7
            ua.u[1] = *(const u32x4*)(ap + 16);     // K = 16+half*8 .. +7
            ap += 32;

            // prefetch next weight strip into registers; latency hidden by WMMAs
            const bool have_next = (it + 1) < niter;
            u32x4 wnext;
            if (have_next) wnext = *(const u32x4*)(wp + 32);
            wp += 32;

            const bf16* bbase = &ldsB[(it & 1) * LDSB_HALVES];
            #pragma unroll
            for (int nt = 0; nt < 4; ++nt) {
                union { u32x4 u[2]; v16bf v; } ub;
                const bf16* bp = bbase + (nt * 16 + row) * LDSB_STRIDE + half * 8;
                ub.u[0] = *(const u32x4*)bp;
                ub.u[1] = *(const u32x4*)(bp + 16);
                acc[nt] = __builtin_amdgcn_wmma_f32_16x16x32_bf16(
                    false, ua.v, false, ub.v, (short)0, acc[nt], false, false);
            }

            if (have_next) {
                // Safe without a pre-barrier: buf[(it+1)&1] was last read at
                // iter it-1, and those ds_loads were consumed (DScnt-waited by
                // the WMMAs) before that iteration's barrier.
                *(u32x4*)&ldsB[((it + 1) & 1) * LDSB_HALVES
                               + jr * LDSB_STRIDE + ks] = wnext;
                __syncthreads();
            }
        }
    }

    #pragma unroll
    for (int nt = 0; nt < 4; ++nt) {
        const int colg = n0 + nt * 16 + row;
        #pragma unroll
        for (int r = 0; r < 8; ++r) {
            const int m = m0 + r + 8 * half;
            Cout[(size_t)m * ldc + colg] = acc[nt][r];
        }
    }
}

// ---------------------------------------------------------------------------
// LSTM pointwise: gates -> h,c ; writes h in fp32 + bf16 (and encoder out)
// ---------------------------------------------------------------------------
__global__ __launch_bounds__(256) void lstm_pointwise(
    const float* __restrict__ gates, float* __restrict__ h, float* __restrict__ c,
    bf16* __restrict__ hbf, bf16* __restrict__ enc_out)
{
    const int idx = blockIdx.x * 256 + threadIdx.x;   // b*H + k, 128*1024 total
    const int b = idx >> 10, k = idx & (CH - 1);
    const float* g = gates + (size_t)b * CG;
    const float gi = g[k], gf = g[k + CH], gg = g[k + 2 * CH], go = g[k + 3 * CH];
    const float si = 1.0f / (1.0f + __expf(-gi));
    const float sf = 1.0f / (1.0f + __expf(-gf));
    const float so = 1.0f / (1.0f + __expf(-go));
    const float cn = sf * c[idx] + si * tanhf(gg);
    const float hn = so * tanhf(cn);
    c[idx] = cn;
    h[idx] = hn;
    const bf16 hb = f2bf(hn);
    hbf[idx] = hb;
    if (enc_out) enc_out[idx] = hb;   // layout [t][b][k], pointer pre-offset by t
}

// ---------------------------------------------------------------------------
// Attention energy + softmax: one block per batch sample.
//   energy[s] = q[b,:] . enc[s,b,:]  (wave per s, lane-parallel over H)
// ---------------------------------------------------------------------------
__global__ __launch_bounds__(256) void attn_energy_softmax(
    const float* __restrict__ q, const bf16* __restrict__ enc, float* __restrict__ wout)
{
    const int b = blockIdx.x;
    const int tid = threadIdx.x, lane = tid & 31, wave = tid >> 5;
    __shared__ float e[CS];
    __shared__ float red[8];

    float qreg[CH / 32];
    const float* qb = q + (size_t)b * CH;
    #pragma unroll
    for (int i = 0; i < CH / 32; ++i) qreg[i] = qb[i * 32 + lane];

    for (int s = wave; s < CS; s += 8) {
        const bf16* ep = enc + ((size_t)s * CB + b) * CH;
        float sum = 0.0f;
        #pragma unroll
        for (int i = 0; i < CH / 32; ++i) sum += qreg[i] * bf2f(ep[i * 32 + lane]);
        for (int off = 16; off > 0; off >>= 1) sum += __shfl_xor(sum, off, 32);
        if (lane == 0) e[s] = sum;
    }
    __syncthreads();

    float m = -3.0e38f;
    for (int s = tid; s < CS; s += 256) m = fmaxf(m, e[s]);
    for (int off = 16; off > 0; off >>= 1) m = fmaxf(m, __shfl_xor(m, off, 32));
    if (lane == 0) red[wave] = m;
    __syncthreads();
    float mm = red[0];
    #pragma unroll
    for (int w = 1; w < 8; ++w) mm = fmaxf(mm, red[w]);
    __syncthreads();

    float sum = 0.0f;
    for (int s = tid; s < CS; s += 256) { float ex = __expf(e[s] - mm); e[s] = ex; sum += ex; }
    for (int off = 16; off > 0; off >>= 1) sum += __shfl_xor(sum, off, 32);
    if (lane == 0) red[wave] = sum;
    __syncthreads();
    float tot = 0.0f;
    #pragma unroll
    for (int w = 0; w < 8; ++w) tot += red[w];
    const float inv = 1.0f / tot;
    for (int s = tid; s < CS; s += 256) wout[(size_t)b * CS + s] = e[s] * inv;
}

// ---------------------------------------------------------------------------
// Attention context: ctx[b,k] = sum_s w[b,s]*enc[s,b,k] -> bf16 (decoder GEMM A)
// ---------------------------------------------------------------------------
__global__ __launch_bounds__(256) void attn_ctx(
    const float* __restrict__ w, const bf16* __restrict__ enc, bf16* __restrict__ ctx)
{
    const int b = blockIdx.x, tid = threadIdx.x;
    float a0 = 0, a1 = 0, a2 = 0, a3 = 0;
    const float* wb = w + (size_t)b * CS;
    for (int s = 0; s < CS; ++s) {
        const float wv = wb[s];
        const bf16* ep = enc + ((size_t)s * CB + b) * CH + tid;
        a0 += wv * bf2f(ep[0]);
        a1 += wv * bf2f(ep[256]);
        a2 += wv * bf2f(ep[512]);
        a3 += wv * bf2f(ep[768]);
    }
    bf16* cp = ctx + (size_t)b * CH + tid;
    cp[0] = f2bf(a0); cp[256] = f2bf(a1); cp[512] = f2bf(a2); cp[768] = f2bf(a3);
}

// ---------------------------------------------------------------------------
// Output projection: pred[b] = fc_W . h[b] + fc_b ; feeds back as prev_y
// ---------------------------------------------------------------------------
__global__ __launch_bounds__(256) void dec_pred(
    const float* __restrict__ h, const float* __restrict__ fcW,
    const float* __restrict__ fcb, float* __restrict__ out,
    float* __restrict__ prev_y, int t)
{
    const int b = blockIdx.x, tid = threadIdx.x, lane = tid & 31, wave = tid >> 5;
    __shared__ float red[8];
    float sum = 0.0f;
    for (int k = tid; k < CH; k += 256) sum += h[(size_t)b * CH + k] * fcW[k];
    for (int off = 16; off > 0; off >>= 1) sum += __shfl_xor(sum, off, 32);
    if (lane == 0) red[wave] = sum;
    __syncthreads();
    if (tid == 0) {
        float s = fcb[0];
        #pragma unroll
        for (int w = 0; w < 8; ++w) s += red[w];
        out[(size_t)b * CT + t] = s;
        prev_y[b] = s;
    }
}

// ---------------------------------------------------------------------------
// Host-side orchestration
// ---------------------------------------------------------------------------
extern "C" void kernel_launch(void* const* d_in, const int* in_sizes, int n_in,
                              void* d_out, int out_size, void* d_ws, size_t ws_size,
                              hipStream_t stream) {
    const float* src      = (const float*)d_in[0];
    const float* encWih   = (const float*)d_in[1];
    const float* encWhh   = (const float*)d_in[2];
    const float* enc_bih  = (const float*)d_in[3];
    const float* enc_bhh  = (const float*)d_in[4];
    const float* decWih   = (const float*)d_in[5];
    const float* decWhh   = (const float*)d_in[6];
    const float* dec_bih  = (const float*)d_in[7];
    const float* dec_bhh  = (const float*)d_in[8];
    const float* attnW    = (const float*)d_in[9];
    const float* fcW      = (const float*)d_in[10];
    const float* fcb      = (const float*)d_in[11];
    float* out = (float*)d_out;

    // workspace carve-up (256B aligned)
    char* ws = (char*)d_ws;
    size_t off = 0;
    auto alloc = [&](size_t bytes) -> void* {
        void* p = ws + off;
        off = (off + bytes + 255) & ~(size_t)255;
        return p;
    };
    bf16*  src_bf   = (bf16*)alloc((size_t)CB * CS * CIN * 2);
    bf16*  encWihB  = (bf16*)alloc((size_t)CG * CIN * 2);
    bf16*  encWhhB  = (bf16*)alloc((size_t)CG * CH * 2);
    bf16*  decWhhB  = (bf16*)alloc((size_t)CG * CH * 2);
    bf16*  decWctxB = (bf16*)alloc((size_t)CG * CH * 2);
    bf16*  attnWB   = (bf16*)alloc((size_t)CH * CH * 2);
    float* wcol0    = (float*)alloc((size_t)CG * 4);
    float* bsum_enc = (float*)alloc((size_t)CG * 4);
    float* bsum_dec = (float*)alloc((size_t)CG * 4);
    float* hS       = (float*)alloc((size_t)CB * CH * 4);
    float* cS       = (float*)alloc((size_t)CB * CH * 4);
    bf16*  hbf      = (bf16*)alloc((size_t)CB * CH * 2);
    float* gates    = (float*)alloc((size_t)CB * CG * 4);
    float* qbuf     = (float*)alloc((size_t)CB * CH * 4);
    float* attn_w   = (float*)alloc((size_t)CB * CS * 4);
    bf16*  ctx_bf   = (bf16*)alloc((size_t)CB * CH * 2);
    float* prev_y   = (float*)alloc((size_t)CB * 4);
    bf16*  enc_out  = (bf16*)alloc((size_t)CS * CB * CH * 2);  // [S][B][H], L2-resident

    auto blocks = [](int n) { return (n + 255) / 256; };

    // ---- one-time packing (re-done every launch: deterministic) ----
    k_f2bf<<<blocks(CB * CS * CIN), 256, 0, stream>>>(src_bf, src, CB * CS * CIN);
    k_f2bf<<<blocks(CG * CIN), 256, 0, stream>>>(encWihB, encWih, CG * CIN);
    k_f2bf<<<blocks(CG * CH), 256, 0, stream>>>(encWhhB, encWhh, CG * CH);
    k_f2bf<<<blocks(CG * CH), 256, 0, stream>>>(decWhhB, decWhh, CG * CH);
    k_f2bf<<<blocks(CH * CH), 256, 0, stream>>>(attnWB, attnW, CH * CH);
    k_pack_decwih<<<blocks(CG * CH), 256, 0, stream>>>(decWctxB, wcol0, decWih);
    k_bias_sum<<<blocks(CG), 256, 0, stream>>>(bsum_enc, enc_bih, enc_bhh, CG);
    k_bias_sum<<<blocks(CG), 256, 0, stream>>>(bsum_dec, dec_bih, dec_bhh, CG);
    k_zero_f<<<blocks(CB * CH), 256, 0, stream>>>(hS, CB * CH);
    k_zero_f<<<blocks(CB * CH), 256, 0, stream>>>(cS, CB * CH);
    k_zero_bf<<<blocks(CB * CH), 256, 0, stream>>>(hbf, CB * CH);
    k_zero_f<<<1, 256, 0, stream>>>(prev_y, CB);

    // ---- encoder scan ----
    for (int t = 0; t < CS; ++t) {
        gates_gemm<<<CG / 64, 256, 0, stream>>>(
            hbf, CH, CH, encWhhB, CH,                 // h @ W_hh^T  (K=1024)
            src_bf + (size_t)t * CIN, CS * CIN, CIN,  // x_t @ W_ih^T (K=64)
            encWihB, CIN,
            bsum_enc, nullptr, nullptr, gates, CG);
        lstm_pointwise<<<blocks(CB * CH), 256, 0, stream>>>(
            gates, hS, cS, hbf, enc_out + (size_t)t * CB * CH);
    }

    // ---- decoder scan (h,c continue from encoder final state) ----
    for (int t = 0; t < CT; ++t) {
        gates_gemm<<<CH / 64, 256, 0, stream>>>(          // q = h @ attn_W^T
            hbf, CH, CH, attnWB, CH,
            nullptr, 0, 0, nullptr, 0,
            nullptr, nullptr, nullptr, qbuf, CH);
        attn_energy_softmax<<<CB, 256, 0, stream>>>(qbuf, enc_out, attn_w);
        attn_ctx<<<CB, 256, 0, stream>>>(attn_w, enc_out, ctx_bf);
        gates_gemm<<<CG / 64, 256, 0, stream>>>(          // dec LSTM gates
            hbf, CH, CH, decWhhB, CH,                     // h @ W_hh^T
            ctx_bf, CH, CH, decWctxB, CH,                 // ctx @ W_ih[:,1:]^T
            bsum_dec, prev_y, wcol0,                      // + prev_y * W_ih[:,0] + bias
            gates, CG);
        lstm_pointwise<<<blocks(CB * CH), 256, 0, stream>>>(
            gates, hS, cS, hbf, nullptr);
        dec_pred<<<CB, 256, 0, stream>>>(hS, fcW, fcb, out, prev_y, t);
    }
}
